// QuantumAttentionHead_65481071399475
// MI455X (gfx1250) — compile-verified, matrix operand order checked by reading, not run on hardware
//
#include <hip/hip_runtime.h>
#include <hip/hip_bf16.h>
#include <stdint.h>

// ---------------------------------------------------------------------------
// QuantumAttentionHead for MI455X (gfx1250, wave32, WMMA)
//   P[b,h,s]  = cos(x[b,s,h])                         (f16, H-major)
//   Pt[b,s,h] = cos(x[b,s,h])                         (f16, S-major == x layout)
//   scores    = P P^T     (2048^3 GEMM, f16 WMMA)     scale = 1/sqrt(1) = 1
//   out       = softmax(scores) P                     (2048^3 GEMM, f16 WMMA)
//   d_out[b,s,h] = out[h,s]
//
// ~69 GFLOP over ~100 MB of traffic -> compute bound; run the 2048^3 GEMMs
// on the f16 WMMA pipe. Dual accumulators per wave give the XDL pipe two
// independent WMMA chains and halve the A-operand LDS traffic.
// ---------------------------------------------------------------------------

typedef __attribute__((ext_vector_type(16))) _Float16 v16h;
typedef __attribute__((ext_vector_type(8)))  _Float16 v8h;
typedef __attribute__((ext_vector_type(8)))  float    v8f;

#define BATCH 2
#define SEQ   2048
#define EMB   2048      // == NUM_HEADS (d_k = 1)
#define HD    2048

static __device__ __forceinline__ v16h pack16(v8h lo, v8h hi) {
    return __builtin_shufflevector(lo, hi, 0,1,2,3,4,5,6,7,8,9,10,11,12,13,14,15);
}

static __device__ __forceinline__ v8f wmma_f16(v16h a, v16h b, v8f c) {
    return __builtin_amdgcn_wmma_f32_16x16x32_f16(false, a, false, b,
                                                  (short)0, c, false, false);
}

// ---------------------------------------------------------------------------
// Kernel 1: elementwise cos + f16 conversion, written in both layouts.
// Memory bound (33 MB read, 34 MB written) -> ~3 us at 23.3 TB/s, negligible.
// ---------------------------------------------------------------------------
__global__ void qah_precompute(const float* __restrict__ x,
                               _Float16* __restrict__ P,    // [B][H][S]
                               _Float16* __restrict__ Pt)   // [B][S][H]
{
    int idx = blockIdx.x * blockDim.x + threadIdx.x;
    const int total = BATCH * SEQ * EMB;
    if (idx >= total) return;
    float v = __cosf(x[idx]);
    _Float16 hv = (_Float16)v;
    Pt[idx] = hv;                                   // same layout as x
    int b = idx / (SEQ * EMB);
    int r = idx - b * (SEQ * EMB);
    int s = r / EMB;
    int h = r - s * EMB;
    P[(size_t)b * HD * SEQ + (size_t)h * SEQ + s] = hv;   // transposed copy
}

// ---------------------------------------------------------------------------
// Kernel 2: fused attention. One block (8 wave32) per (batch, 16-row q tile).
// ---------------------------------------------------------------------------
__global__ void __launch_bounds__(256)
qah_attention(const _Float16* __restrict__ P,
              const _Float16* __restrict__ Pt,
              float* __restrict__ out)
{
    __shared__ _Float16 smemQ[16 * SEQ];   // 64 KB: staged Q tile (rows h0..h0+15)
    __shared__ _Float16 smemS[16 * HD];    // 64 KB: scores, then exp(scores - max)
    __shared__ float    red[16 * 16];
    __shared__ float    rowmax[16];
    __shared__ float    rowinv[16];

    const int tid  = threadIdx.x;
    const int lane = tid & 31;
    const int wave = tid >> 5;                 // 0..7
    const int b    = blockIdx.x >> 7;          // 128 q-tiles per batch
    const int qt   = blockIdx.x & 127;
    const int h0   = qt * 16;

    const _Float16* Pb  = P  + (size_t)b * HD  * SEQ;
    const _Float16* Ptb = Pt + (size_t)b * SEQ * HD;

    // ---- stage Q tile (16 rows x 2048 f16 = one contiguous 64 KB span) into
    //      LDS with CDNA5 async global->LDS DMA (tracked by ASYNCcnt).
    {
        const uint64_t gbase = (uint64_t)(uintptr_t)(Pb + (size_t)h0 * SEQ);
        const uint32_t lbase = (uint32_t)(uintptr_t)(&smemQ[0]);
        #pragma unroll
        for (int i = 0; i < 16; ++i) {
            uint32_t loff = lbase + (uint32_t)(tid * 16 + i * 4096);
            uint64_t goff = gbase + (uint64_t)(tid * 16 + i * 4096);
            asm volatile("global_load_async_to_lds_b128 %0, %1, off"
                         :: "v"(loff), "v"(goff) : "memory");
        }
        asm volatile("s_wait_asynccnt 0x0" ::: "memory");
    }
    __syncthreads();

    // Per the 16-bit A/B WMMA operand layout (ISA 7.12.2):
    //   lanes 0-15  hold M/N = lane,    K chunks {k..k+7} and {k+16..k+23}
    //   lanes 16-31 hold M/N = lane-16, K chunks {k+8..k+15} and {k+24..k+31}
    const int nm = lane & 15;                  // A row == B column for our tiles
    const int kh = (lane >> 4) << 3;           // 0 or 8
    const int mb = (lane >> 4) << 3;           // C/D row base per lane half

    // ======== GEMM1: scores[16 x 2048] = Q . K^T, 2 key tiles / iter =======
    for (int gt = wave * 2; gt < 128; gt += 16) {
        const _Float16* Krow0 = Pb + (size_t)(gt * 16 + nm) * SEQ;
        const _Float16* Krow1 = Krow0 + 16 * SEQ;
        __builtin_prefetch(Krow0, 0, 3);       // global_prefetch_b8
        __builtin_prefetch(Krow1, 0, 3);
        v8f c0 = {};
        v8f c1 = {};
        #pragma unroll 4
        for (int ks = 0; ks < 64; ++ks) {
            const int k = ks * 32 + kh;
            v8h alo = *(const v8h*)&smemQ[nm * SEQ + k];
            v8h ahi = *(const v8h*)&smemQ[nm * SEQ + k + 16];
            v16h a  = pack16(alo, ahi);
            v8h b0l = *(const v8h*)&Krow0[k];
            v8h b0h = *(const v8h*)&Krow0[k + 16];
            v8h b1l = *(const v8h*)&Krow1[k];
            v8h b1h = *(const v8h*)&Krow1[k + 16];
            c0 = wmma_f16(a, pack16(b0l, b0h), c0);   // two independent
            c1 = wmma_f16(a, pack16(b1l, b1h), c1);   // accumulator chains
        }
        // scale = 1/sqrt(d_k) = 1.0 -> nothing to apply
        #pragma unroll
        for (int j = 0; j < 8; ++j) {
            smemS[(mb + j) * HD + gt * 16 + nm]       = (_Float16)c0[j];
            smemS[(mb + j) * HD + (gt + 1) * 16 + nm] = (_Float16)c1[j];
        }
    }
    __syncthreads();

    // ================= softmax (unnormalized; fold 1/sum into epilogue) ====
    {
        const int row = tid & 15;
        const int seg = tid >> 4;              // 16 threads per row
        const int base = row * HD + seg * 128;
        float lmax = -3.0e38f;
        for (int i = 0; i < 128; ++i)
            lmax = fmaxf(lmax, (float)smemS[base + i]);
        red[row * 16 + seg] = lmax;
        __syncthreads();
        if (tid < 16) {
            float mx = red[tid * 16];
            for (int i = 1; i < 16; ++i) mx = fmaxf(mx, red[tid * 16 + i]);
            rowmax[tid] = mx;
        }
        __syncthreads();
        const float mx = rowmax[row];
        float lsum = 0.f;
        for (int i = 0; i < 128; ++i) {
            float e = __expf((float)smemS[base + i] - mx);
            lsum += e;
            smemS[base + i] = (_Float16)e;     // store exp(s - max) as f16
        }
        red[row * 16 + seg] = lsum;
        __syncthreads();
        if (tid < 16) {
            float s = 0.f;
            for (int i = 0; i < 16; ++i) s += red[tid * 16 + i];
            rowinv[tid] = 1.0f / s;
        }
        __syncthreads();
    }

    // ======== GEMM2: out[16 x 2048] = probs . P, 2 s-tiles / iter ==========
    // B[k,n] = P[g0+k, s0+n] = Pt[s0+n][g0+k]  -> contiguous in K per lane.
    for (int st = wave * 2; st < 128; st += 16) {
        const _Float16* Vrow0 = Ptb + (size_t)(st * 16 + nm) * HD;
        const _Float16* Vrow1 = Vrow0 + 16 * HD;
        __builtin_prefetch(Vrow0, 0, 3);
        __builtin_prefetch(Vrow1, 0, 3);
        v8f c0 = {};
        v8f c1 = {};
        #pragma unroll 4
        for (int ks = 0; ks < 64; ++ks) {
            const int k = ks * 32 + kh;
            v8h alo = *(const v8h*)&smemS[nm * HD + k];
            v8h ahi = *(const v8h*)&smemS[nm * HD + k + 16];
            v16h a  = pack16(alo, ahi);
            v8h b0l = *(const v8h*)&Vrow0[k];
            v8h b0h = *(const v8h*)&Vrow0[k + 16];
            v8h b1l = *(const v8h*)&Vrow1[k];
            v8h b1h = *(const v8h*)&Vrow1[k + 16];
            c0 = wmma_f16(a, pack16(b0l, b0h), c0);
            c1 = wmma_f16(a, pack16(b1l, b1h), c1);
        }
        // D rows mb..mb+7 are consecutive h indices -> two float4 stores per
        // tile; transposed write: d_out[b, s0+nm, h0+mb+j] = c[j] / rowsum
        float* orow0 = out + (size_t)b * SEQ * EMB
                           + (size_t)(st * 16 + nm) * EMB + h0 + mb;
        float* orow1 = orow0 + (size_t)16 * EMB;
        float4 a0 = make_float4(c0[0] * rowinv[mb + 0], c0[1] * rowinv[mb + 1],
                                c0[2] * rowinv[mb + 2], c0[3] * rowinv[mb + 3]);
        float4 a1 = make_float4(c0[4] * rowinv[mb + 4], c0[5] * rowinv[mb + 5],
                                c0[6] * rowinv[mb + 6], c0[7] * rowinv[mb + 7]);
        float4 b0 = make_float4(c1[0] * rowinv[mb + 0], c1[1] * rowinv[mb + 1],
                                c1[2] * rowinv[mb + 2], c1[3] * rowinv[mb + 3]);
        float4 b1 = make_float4(c1[4] * rowinv[mb + 4], c1[5] * rowinv[mb + 5],
                                c1[6] * rowinv[mb + 6], c1[7] * rowinv[mb + 7]);
        *(float4*)(orow0)     = a0;
        *(float4*)(orow0 + 4) = a1;
        *(float4*)(orow1)     = b0;
        *(float4*)(orow1 + 4) = b1;
    }
}

// ---------------------------------------------------------------------------
extern "C" void kernel_launch(void* const* d_in, const int* in_sizes, int n_in,
                              void* d_out, int out_size, void* d_ws, size_t ws_size,
                              hipStream_t stream) {
    const float* x = (const float*)d_in[0];
    float* o = (float*)d_out;

    const size_t nElem = (size_t)BATCH * SEQ * EMB;        // 8.4M
    _Float16* P  = (_Float16*)d_ws;                        // 16.8 MB
    _Float16* Pt = (_Float16*)((char*)d_ws + nElem * sizeof(_Float16));

    qah_precompute<<<(int)((nElem + 255) / 256), 256, 0, stream>>>(x, P, Pt);
    qah_attention<<<BATCH * (HD / 16), 256, 0, stream>>>(P, Pt, o);
}
